// NonLocalBlock_36593121361934
// MI455X (gfx1250) — matrix operand order repaired and checked by reading, not compiled, not run on hardware
//
#include <hip/hip_runtime.h>
#include <hip/hip_bf16.h>

typedef __attribute__((ext_vector_type(16))) _Float16 v16h;
typedef __attribute__((ext_vector_type(8)))  float    v8f;
typedef __attribute__((ext_vector_type(4)))  int      v4i;
typedef __attribute__((ext_vector_type(4)))  unsigned u32x4;
typedef __attribute__((ext_vector_type(8)))  int      i32x8;
typedef __attribute__((ext_vector_type(4)))  int      i32x4;

union AF { v16h h; v4i q[2]; };

#define C_DIM 512
#define N_SP  4096
#define B_DIM 4
#define NGRP  32
#define CPG   16          // channels per group
#define GN_ELEMS (CPG * N_SP)   // 65536 per (b,g)

static __device__ __forceinline__ v8f wmma_f16(const v16h& a, const v16h& b, const v8f& c) {
    return __builtin_amdgcn_wmma_f32_16x16x32_f16(false, a, false, b, (short)0, c, false, false);
}

// ---- CDNA5 data-mover paths, layered: TDM > async-to-LDS > plain copies ----
#if __has_builtin(__builtin_amdgcn_tensor_load_to_lds)
#define HAVE_TDM 1
#endif
#if __has_builtin(__builtin_amdgcn_global_load_async_to_lds_b128)
#define HAVE_ASYNC_LDS 1
typedef __attribute__((address_space(1))) v4i as1_v4i;
typedef __attribute__((address_space(3))) v4i as3_v4i;
#endif

static __device__ __forceinline__ void cp16_lds(void* lds, const void* g) {
#ifdef HAVE_ASYNC_LDS
    __builtin_amdgcn_global_load_async_to_lds_b128((as1_v4i*)g, (as3_v4i*)lds, 0, 0);
#else
    *(v4i*)lds = *(const v4i*)g;
#endif
}

#ifdef HAVE_TDM
static __device__ __forceinline__ unsigned lds_off_of(const void* p) {
    return (unsigned)(size_t)(__attribute__((address_space(3))) const void*)p;
}

// 2D TDM tile load: f16 tensor [dim1][dim0] row-major (stride0 elements per
// row), tile [tile1][tile0] starting at gaddr, packed contiguously at ldsoff.
static __device__ __forceinline__ void tdm_load_2d(
        const void* gaddr, unsigned ldsoff,
        unsigned dim0, unsigned dim1, unsigned tile0, unsigned tile1,
        unsigned stride0) {
    unsigned long long ga = (unsigned long long)(size_t)gaddr;
    u32x4 g0;
    g0[0] = 1u;                                   // count=1, user mode
    g0[1] = ldsoff;                               // lds_addr [63:32]
    g0[2] = (unsigned)(ga & 0xffffffffu);         // global_addr [95:64]
    g0[3] = (unsigned)((ga >> 32) & 0x1ffffffu)   // global_addr [120:96]
          | (2u << 30);                           // type=2 (image)
    i32x8 g1;
    g1[0] = (int)(1u << 16);                      // data_size=1 -> 2 bytes
    g1[1] = (int)((dim0 & 0xffffu) << 16);        // tensor_dim0 [79:48] lo
    g1[2] = (int)((dim0 >> 16) | ((dim1 & 0xffffu) << 16));  // dim0 hi | dim1 lo
    g1[3] = (int)((dim1 >> 16) | (tile0 << 16));  // dim1 hi | tile_dim0
    g1[4] = (int)(tile1 & 0xffffu);               // tile_dim1 | tile_dim2=0
    g1[5] = (int)stride0;                         // tensor_dim0_stride lo32
    g1[6] = 0;                                    // stride0 hi | stride1 lo
    g1[7] = 0;                                    // stride1 hi
    i32x4 z4 = {0, 0, 0, 0};                      // groups 2/3 unused (2D)
    i32x8 z8 = {0, 0, 0, 0, 0, 0, 0, 0};          // extra group (6-arg form)
    __builtin_amdgcn_tensor_load_to_lds(g0, g1, z4, z4, z8, 0);
}
#endif

static __device__ __forceinline__ void wait_tiles() {
#ifdef HAVE_TDM
#if __has_builtin(__builtin_amdgcn_s_wait_tensorcnt)
    __builtin_amdgcn_s_wait_tensorcnt(0);
#else
    asm volatile("s_wait_tensorcnt 0" ::: "memory");
#endif
#elif defined(HAVE_ASYNC_LDS)
#if __has_builtin(__builtin_amdgcn_s_wait_asynccnt)
    __builtin_amdgcn_s_wait_asynccnt(0);
#else
    asm volatile("s_wait_asynccnt 0" ::: "memory");
#endif
#endif
}

// ---------------------------------------------------------------------------
// Kernel 1: GroupNorm statistics. 128 blocks (b*32+g), 256 threads.
// ---------------------------------------------------------------------------
__global__ void gn_stats_kernel(const float* __restrict__ x, float* __restrict__ stats) {
    int bg = blockIdx.x;                       // 0..127
    const float4* p = (const float4*)(x + (size_t)bg * GN_ELEMS);
    float s = 0.f, ss = 0.f;
    for (int i = threadIdx.x; i < GN_ELEMS / 4; i += 256) {
        float4 v = p[i];
        s  += v.x + v.y + v.z + v.w;
        ss += v.x * v.x + v.y * v.y + v.z * v.z + v.w * v.w;
    }
    __shared__ float rs[256], rss[256];
    rs[threadIdx.x] = s; rss[threadIdx.x] = ss;
    __syncthreads();
    for (int off = 128; off > 0; off >>= 1) {
        if (threadIdx.x < off) {
            rs[threadIdx.x]  += rs[threadIdx.x + off];
            rss[threadIdx.x] += rss[threadIdx.x + off];
        }
        __syncthreads();
    }
    if (threadIdx.x == 0) {
        float m   = rs[0] * (1.0f / GN_ELEMS);
        float var = rss[0] * (1.0f / GN_ELEMS) - m * m;
        stats[bg]       = m;
        stats[128 + bg] = rsqrtf(var + 1e-6f);
    }
}

// ---------------------------------------------------------------------------
// Kernel 2: normalize + affine + cast f16, transpose [b][c][n] -> h16[b*N+n][c]
// ---------------------------------------------------------------------------
__global__ void gn_apply_kernel(const float* __restrict__ x,
                                const float* __restrict__ gamma,
                                const float* __restrict__ beta,
                                const float* __restrict__ stats,
                                _Float16* __restrict__ h16) {
    size_t tid = (size_t)blockIdx.x * 256 + threadIdx.x;   // B*C*N/4 threads
    int n4 = (int)(tid % (N_SP / 4));
    int c  = (int)((tid / (N_SP / 4)) % C_DIM);
    int b  = (int)(tid / ((size_t)(N_SP / 4) * C_DIM));
    int bg = b * NGRP + (c >> 4);
    float m = stats[bg], r = stats[128 + bg];
    float ga = gamma[c] * r;
    float be = beta[c] - m * ga;
    float4 xv = *(const float4*)(x + ((size_t)(b * C_DIM + c)) * N_SP + n4 * 4);
    _Float16* dst = h16 + ((size_t)b * N_SP + (size_t)n4 * 4) * C_DIM + c;
    dst[0 * C_DIM] = (_Float16)(xv.x * ga + be);
    dst[1 * C_DIM] = (_Float16)(xv.y * ga + be);
    dst[2 * C_DIM] = (_Float16)(xv.z * ga + be);
    dst[3 * C_DIM] = (_Float16)(xv.w * ga + be);
}

// ---------------------------------------------------------------------------
// Kernel 3: convert weights to f16; fold 1/sqrt(C) into wq.
// ---------------------------------------------------------------------------
__global__ void wconv_kernel(const float* __restrict__ wq,
                             const float* __restrict__ wk,
                             const float* __restrict__ wv,
                             _Float16* __restrict__ w16) {
    const float scale = 0.04419417382415922f;  // 512^-0.5
    int i = blockIdx.x * 256 + threadIdx.x;    // 0..262143
    w16[i]              = (_Float16)(wq[i] * scale);
    w16[262144 + i]     = (_Float16)wk[i];
    w16[2 * 262144 + i] = (_Float16)wv[i];
}

// ---------------------------------------------------------------------------
// Kernel 4: projection GEMM.  out[m][co] = sum_c h16[m][c] * w[co][c] + bias
// grid (256 mtiles, 4 ntiles, 3 matrices), 128 threads (4 waves).
// Q,K written row-major [m][co]; V written transposed [b][co][n].
// ---------------------------------------------------------------------------
__global__ __launch_bounds__(128) void proj_gemm_kernel(
        const _Float16* __restrict__ h16, const _Float16* __restrict__ w16,
        const float* __restrict__ bq, const float* __restrict__ bk,
        const float* __restrict__ bv_,
        _Float16* __restrict__ q16, _Float16* __restrict__ k16,
        _Float16* __restrict__ v16) {
    int mat  = blockIdx.z;
    int lane = threadIdx.x & 31;
    int wave = threadIdx.x >> 5;
    int rowBase = blockIdx.x * 64 + wave * 16;
    int colBase = blockIdx.y * 128;
    int l15   = lane & 15;
    int akoff = (lane < 16) ? 0 : 8;    // A-operand K base per half
    int bkoff = (lane < 16) ? 0 : 8;    // B-operand K base per half

    const _Float16* wmat = w16 + (size_t)mat * 262144;
    v8f acc[8];
#pragma unroll
    for (int t = 0; t < 8; ++t)
#pragma unroll
        for (int e = 0; e < 8; ++e) acc[t][e] = 0.f;

#pragma unroll 4
    for (int kc = 0; kc < 16; ++kc) {
        AF a;
        const _Float16* ap = h16 + (size_t)(rowBase + l15) * C_DIM + kc * 32 + akoff;
        a.q[0] = *(const v4i*)ap;
        a.q[1] = *(const v4i*)(ap + 16);
#pragma unroll
        for (int nt = 0; nt < 8; ++nt) {
            AF bm;
            const _Float16* bp = wmat + (size_t)(colBase + nt * 16 + l15) * C_DIM + kc * 32 + bkoff;
            bm.q[0] = *(const v4i*)bp;
            bm.q[1] = *(const v4i*)(bp + 16);
            acc[nt] = wmma_f16(a.h, bm.h, acc[nt]);
        }
    }

    const float  bscale = (mat == 0) ? 0.04419417382415922f : 1.0f;
    const float* bias   = (mat == 0) ? bq : (mat == 1) ? bk : bv_;
    int mAdd = (lane < 16) ? 0 : 8;
#pragma unroll
    for (int nt = 0; nt < 8; ++nt) {
        int col = colBase + nt * 16 + l15;
        float bsv = bias[col] * bscale;
#pragma unroll
        for (int r = 0; r < 8; ++r) {
            int m = rowBase + r + mAdd;
            float val = acc[nt][r] + bsv;
            if (mat == 0) {
                q16[(size_t)m * C_DIM + col] = (_Float16)val;
            } else if (mat == 1) {
                k16[(size_t)m * C_DIM + col] = (_Float16)val;
            } else {
                int b = m >> 12, n = m & 4095;
                v16[((size_t)b * C_DIM + col) * N_SP + n] = (_Float16)val;
            }
        }
    }
}

// ---------------------------------------------------------------------------
// Kernel 5: fused flash attention + residual, double-buffered TDM tiles.
// grid = B * (N/64) = 256 blocks, 256 threads (8 waves).
// wave = rgrp (0..3: 16 query rows) x cg (0..1: 256 output channels).
// LDS: 2x K tile (32x512 f16) + 2x V tile (512x32 f16) + P staging = 136 KB.
// Wave 0 issues two tensor_load_to_lds per iteration (next tile) while all
// waves run 48 WMMAs on the current tile; s_wait_tensorcnt + barrier publish.
// ---------------------------------------------------------------------------
__global__ __launch_bounds__(256, 1) void attn_kernel(
        const _Float16* __restrict__ q16, const _Float16* __restrict__ k16,
        const _Float16* __restrict__ v16, const float* __restrict__ x,
        float* __restrict__ out) {
    __shared__ _Float16 Kbuf[2][32 * C_DIM];   // 2 x 32 KB
    __shared__ _Float16 Vbuf[2][C_DIM * 32];   // 2 x 32 KB
    __shared__ _Float16 Pbuf[8][16 * 32];      // 8 KB, wave-private staging

    int b     = blockIdx.x >> 6;
    int qtile = blockIdx.x & 63;
    int lane  = threadIdx.x & 31;
    int wave  = threadIdx.x >> 5;
    int rgrp  = wave & 3;
    int cg    = wave >> 2;
    int l15   = lane & 15;
    int akoff = (lane < 16) ? 0 : 8;
    int bkoff = (lane < 16) ? 0 : 8;
    int mAdd  = (lane < 16) ? 0 : 8;

    int rowBase = qtile * 64 + rgrp * 16;   // spatial index of first query row
    const _Float16* Abase = q16 + ((size_t)b * N_SP + rowBase + l15) * C_DIM + akoff;
    const _Float16* kb_g  = k16 + (size_t)b * N_SP * C_DIM;
    const _Float16* vb_g  = v16 + (size_t)b * C_DIM * N_SP;
    _Float16* Pst = &Pbuf[wave][0];

    v8f O[16];
#pragma unroll
    for (int t = 0; t < 16; ++t)
#pragma unroll
        for (int e = 0; e < 8; ++e) O[t][e] = 0.f;
    float rmax[8], rsum[8];
#pragma unroll
    for (int r = 0; r < 8; ++r) { rmax[r] = -__builtin_inff(); rsum[r] = 0.f; }

    // tile loader: K tile 32 keys x 512 ch (row-major), V tile 512 ch x 32 keys
    auto load_tiles = [&](int jt, int bsel) {
#ifdef HAVE_TDM
        if (wave == 0) {
            // K: tensor [4096][512] f16, tile rows jt*32..+31, full width
            tdm_load_2d(kb_g + (size_t)jt * 32 * C_DIM, lds_off_of(&Kbuf[bsel][0]),
                        C_DIM, N_SP, C_DIM, 32, C_DIM);
            // V^T: tensor [512][4096] f16, tile cols jt*32..+31, full height
            tdm_load_2d(vb_g + (size_t)jt * 32, lds_off_of(&Vbuf[bsel][0]),
                        N_SP, C_DIM, 32, C_DIM, N_SP);
        }
#else
        const v4i* src = (const v4i*)(kb_g + (size_t)jt * 32 * C_DIM);
        v4i* kdst = (v4i*)&Kbuf[bsel][0];
#pragma unroll
        for (int i = 0; i < 8; ++i) {
            int idx = threadIdx.x + 256 * i;
            cp16_lds(kdst + idx, src + idx);
        }
        v4i* vdst = (v4i*)&Vbuf[bsel][0];
#pragma unroll
        for (int i = 0; i < 8; ++i) {
            int idx = threadIdx.x + 256 * i;
            int c = idx >> 2, seg = idx & 3;
            cp16_lds(vdst + idx, vb_g + (size_t)c * N_SP + jt * 32 + seg * 8);
        }
#endif
    };

    load_tiles(0, 0);

    for (int jt = 0; jt < N_SP / 32; ++jt) {
        int cur = jt & 1;
        wait_tiles();         // issuing wave(s): buf[cur] transfers landed
        __syncthreads();      // tile published to all waves

        if (jt + 1 < N_SP / 32) load_tiles(jt + 1, cur ^ 1);

        const _Float16* Kt = &Kbuf[cur][0];
        const _Float16* Vt = &Vbuf[cur][0];

        // S = Q . K^T  (16 queries x 32 keys), reduction over C=512
        v8f s0, s1;
#pragma unroll
        for (int e = 0; e < 8; ++e) { s0[e] = 0.f; s1[e] = 0.f; }
#pragma unroll
        for (int kc = 0; kc < 16; ++kc) {
            AF a;
            const _Float16* ap = Abase + kc * 32;
            a.q[0] = *(const v4i*)ap;
            a.q[1] = *(const v4i*)(ap + 16);
            const _Float16* bp0 = Kt + (size_t)l15 * C_DIM + kc * 32 + bkoff;
            AF b0, b1;
            b0.q[0] = *(const v4i*)bp0;
            b0.q[1] = *(const v4i*)(bp0 + 16);
            const _Float16* bp1 = bp0 + 16 * C_DIM;
            b1.q[0] = *(const v4i*)bp1;
            b1.q[1] = *(const v4i*)(bp1 + 16);
            s0 = wmma_f16(a.h, b0.h, s0);
            s1 = wmma_f16(a.h, b1.h, s1);
        }

        // online softmax (row stats live per 16-lane half == per query row)
        float fac[8];
#pragma unroll
        for (int r = 0; r < 8; ++r) {
            float tm = fmaxf(s0[r], s1[r]);
            tm = fmaxf(tm, __shfl_xor(tm, 1, 32));
            tm = fmaxf(tm, __shfl_xor(tm, 2, 32));
            tm = fmaxf(tm, __shfl_xor(tm, 4, 32));
            tm = fmaxf(tm, __shfl_xor(tm, 8, 32));
            float nm = fmaxf(rmax[r], tm);
            fac[r] = __expf(rmax[r] - nm);
            float e0 = __expf(s0[r] - nm);
            float e1 = __expf(s1[r] - nm);
            float ts = e0 + e1;
            ts += __shfl_xor(ts, 1, 32);
            ts += __shfl_xor(ts, 2, 32);
            ts += __shfl_xor(ts, 4, 32);
            ts += __shfl_xor(ts, 8, 32);
            rsum[r] = rsum[r] * fac[r] + ts;
            rmax[r] = nm;
            s0[r] = e0; s1[r] = e1;   // s now holds P (f32)
        }
        // rescale running O
#pragma unroll
        for (int t = 0; t < 16; ++t)
#pragma unroll
            for (int r = 0; r < 8; ++r) O[t][r] *= fac[r];

        // stage P row-major 16x32 (C/D layout -> memory), reload in A layout;
        // Pbuf is wave-private, DS ops are in-order per wave: no barrier needed
#pragma unroll
        for (int r = 0; r < 8; ++r) {
            int m = r + mAdd;
            Pst[m * 32 + l15]      = (_Float16)s0[r];
            Pst[m * 32 + 16 + l15] = (_Float16)s1[r];
        }
        AF pa;
        const _Float16* pp = Pst + l15 * 32 + akoff;
        pa.q[0] = *(const v4i*)pp;
        pa.q[1] = *(const v4i*)(pp + 16);

        // O += P @ V  (this wave's 256-channel slice = 16 tiles)
#pragma unroll
        for (int ct = 0; ct < 16; ++ct) {
            const _Float16* vp = Vt + (size_t)(cg * 256 + ct * 16 + l15) * 32 + bkoff;
            AF bvv;
            bvv.q[0] = *(const v4i*)vp;
            bvv.q[1] = *(const v4i*)(vp + 16);
            O[ct] = wmma_f16(pa.h, bvv.h, O[ct]);
        }
    }

    // epilogue: normalize, residual, store fp32
    float inv[8];
#pragma unroll
    for (int r = 0; r < 8; ++r) inv[r] = 1.0f / rsum[r];
    const float* xb = x + (size_t)b * C_DIM * N_SP;
    float* ob = out + (size_t)b * C_DIM * N_SP;
#pragma unroll
    for (int ct = 0; ct < 16; ++ct) {
        int c = cg * 256 + ct * 16 + l15;
#pragma unroll
        for (int r = 0; r < 8; ++r) {
            int n = rowBase + r + mAdd;
            size_t a = (size_t)c * N_SP + n;
            ob[a] = O[ct][r] * inv[r] + xb[a];
        }
    }
}

// ---------------------------------------------------------------------------
// Launch
// ---------------------------------------------------------------------------
extern "C" void kernel_launch(void* const* d_in, const int* in_sizes, int n_in,
                              void* d_out, int out_size, void* d_ws, size_t ws_size,
                              hipStream_t stream) {
    const float* x     = (const float*)d_in[0];
    const float* gamma = (const float*)d_in[1];
    const float* beta  = (const float*)d_in[2];
    const float* wq    = (const float*)d_in[3];
    const float* bq    = (const float*)d_in[4];
    const float* wk    = (const float*)d_in[5];
    const float* bk    = (const float*)d_in[6];
    const float* wv    = (const float*)d_in[7];
    const float* bv    = (const float*)d_in[8];
    float* out = (float*)d_out;

    char* ws = (char*)d_ws;
    float*    stats = (float*)ws;                              // 1 KB
    _Float16* w16   = (_Float16*)(ws + 4096);                  // 1.5 MB
    _Float16* h16   = (_Float16*)(ws + (size_t)2  * 1024 * 1024);  // 16 MB
    _Float16* q16   = (_Float16*)(ws + (size_t)18 * 1024 * 1024);  // 16 MB
    _Float16* k16   = (_Float16*)(ws + (size_t)34 * 1024 * 1024);  // 16 MB
    _Float16* v16   = (_Float16*)(ws + (size_t)50 * 1024 * 1024);  // 16 MB

    gn_stats_kernel<<<B_DIM * NGRP, 256, 0, stream>>>(x, stats);
    gn_apply_kernel<<<(B_DIM * C_DIM * N_SP / 4) / 256, 256, 0, stream>>>(
        x, gamma, beta, stats, h16);
    wconv_kernel<<<262144 / 256, 256, 0, stream>>>(wq, wk, wv, w16);
    proj_gemm_kernel<<<dim3(256, 4, 3), 128, 0, stream>>>(
        h16, w16, bq, bk, bv, q16, k16, v16);
    attn_kernel<<<B_DIM * (N_SP / 64), 256, 0, stream>>>(q16, k16, v16, x, out);
}
